// MultiHeadedAttention_82257213653587
// MI455X (gfx1250) — matrix operand order repaired and checked by reading, not compiled, not run on hardware
//
#include <hip/hip_runtime.h>

#define BB 4
#define TT 4096
#define CC 64
#define HH 4
#define DD 16

typedef __attribute__((ext_vector_type(16))) __bf16         v16bf;
typedef __attribute__((ext_vector_type(8)))  float          v8f;
typedef __attribute__((ext_vector_type(16))) unsigned short v16us;
typedef __attribute__((ext_vector_type(8)))  unsigned short v8us;

// f32 -> bf16 round-to-nearest-even (bit-level, toolchain-safe)
static __device__ __forceinline__ unsigned short f2bf(float f) {
  unsigned u = __builtin_bit_cast(unsigned, f);
  u += 0x7FFFu + ((u >> 16) & 1u);
  return (unsigned short)(u >> 16);
}
// gather alpha / l of a given query row from its owner lane (0..15)
static __device__ __forceinline__ float bcast_lane(float v, int src) {
  return __builtin_bit_cast(float,
      __builtin_amdgcn_ds_bpermute(src << 2, __builtin_bit_cast(int, v)));
}
// partner-lane swap (lane ^ 16): group-of-32 swizzle, xor=0x10, and=0x1F
static __device__ __forceinline__ float swap16(float v) {
  return __builtin_bit_cast(float,
      __builtin_amdgcn_ds_swizzle(__builtin_bit_cast(int, v), 0x401F));
}

// ---------------- Phase 1: projections q,k (bf16 rows) and V^T (bf16) -------
__global__ __launch_bounds__(256) void proj_kernel(
    const float* __restrict__ x, const float* __restrict__ Wq,
    const float* __restrict__ Wk, const float* __restrict__ Wv,
    unsigned short* __restrict__ qb, unsigned short* __restrict__ kb,
    unsigned short* __restrict__ vt) {
  int z  = blockIdx.x * 256 + threadIdx.x;   // [0, B*H*T)
  int t  = z & (TT - 1);
  int bh = z >> 12;                          // b*H + h (uniform per block)
  int h  = bh & (HH - 1);
  int b  = bh >> 2;

  const float* xr = x + (size_t)(b * TT + t) * CC;
  float xv[CC];
#pragma unroll
  for (int c = 0; c < CC; ++c) xv[c] = xr[c];

  for (int d = 0; d < DD; ++d) {
    const float* wq = Wq + (h * DD + d) * CC;   // uniform -> scalar loads
    const float* wk = Wk + (h * DD + d) * CC;
    float aq = 0.f, ak = 0.f;
#pragma unroll
    for (int c = 0; c < CC; ++c) { aq += xv[c] * wq[c]; ak += xv[c] * wk[c]; }
    qb[((size_t)bh * TT + t) * DD + d] = f2bf(aq);
    kb[((size_t)bh * TT + t) * DD + d] = f2bf(ak);
  }
  for (int e = 0; e < CC; ++e) {
    const float* wv = Wv + (h * CC + e) * CC;
    float av = 0.f;
#pragma unroll
    for (int c = 0; c < CC; ++c) av += xv[c] * wv[c];
    vt[((size_t)bh * CC + e) * TT + t] = f2bf(av);  // transposed, coalesced
  }
}

// ---------------- Phase 2: causal flash attention, bf16 WMMA ----------------
// One wave = 16 query rows. Per 32-key step:
//   S^T tiles = wmma(K_tile, Q^T)  -> lands directly in P's A-operand layout
//   online softmax in exp2 domain, P -> bf16, O += wmma(P, V_tile) x4 tiles
// Loop is peeled: unmasked main loop + exactly one masked tail block.
__global__ __launch_bounds__(256) void fa_kernel(
    const unsigned short* __restrict__ qb, const unsigned short* __restrict__ kb,
    const unsigned short* __restrict__ vt, float* __restrict__ hout) {
  const int tileq = blockIdx.x;              // 0..T/128-1
  const int h = blockIdx.y, b = blockIdx.z;
  const int bh   = b * HH + h;
  const int wave = threadIdx.x >> 5;
  const int lane = threadIdx.x & 31;
  const int q0   = tileq * 128 + wave * 16;
  const int col  = lane & 15;
  const bool hi  = lane >= 16;

  const unsigned short* qrow  = qb + (size_t)bh * TT * DD;
  const unsigned short* krow  = kb + (size_t)bh * TT * DD;
  const unsigned short* vbase = vt + (size_t)bh * CC * TT;

  // Q as B operand (contraction dim 16, zero-padded to 32):
  // lanes 0..15 = full 16-dim q vector of query q0+lane; lanes 16..31 = 0
  v16us qus = {};
  if (!hi) qus = *(const v16us*)(qrow + (size_t)(q0 + lane) * DD);
  v16bf qB = __builtin_bit_cast(v16bf, qus);

  v8f o0 = {}, o1 = {}, o2 = {}, o3 = {};
  const v8f zero8 = {};
  float m = -__builtin_inff(), l = 0.f;
  const int q   = q0 + col;                   // this lane's query row
  const int dof = hi ? 8 : 0;                 // key offset of this lane half (S^T rows, V's k)
  // scores kept in exp2 domain: s * (1/4) * log2(e)
  const float SC = 0.25f * 1.44269504088896340736f;

  auto step = [&](int j, bool masked) __attribute__((always_inline)) {
    // K rows as A operands (contiguous k order: full 16-dim row per lane);
    // upper lane half is multiplied by zero Q lanes, so it just reloads the (finite) row.
    v16us k1 = *(const v16us*)(krow + (size_t)(j + col) * DD);
    v16us k2 = *(const v16us*)(krow + (size_t)(j + 16 + col) * DD);
    v16bf a1 = __builtin_bit_cast(v16bf, k1);
    v16bf a2 = __builtin_bit_cast(v16bf, k2);

    // S^T = K * Q^T : lane holds its query's scores for 16 of 32 keys
    v8f s1 = __builtin_amdgcn_wmma_f32_16x16x32_bf16(
        false, a1, false, qB, (short)0, zero8, false, false);
    v8f s2 = __builtin_amdgcn_wmma_f32_16x16x32_bf16(
        false, a2, false, qB, (short)0, zero8, false, false);

    float e[16];
    float rmax = -__builtin_inff();
#pragma unroll
    for (int r = 0; r < 8; ++r) {
      float v1 = s1[r] * SC;
      float v2 = s2[r] * SC;
      if (masked) {                            // branchless cndmask selects
        v1 = (j + dof + r > q)      ? -__builtin_inff() : v1;
        v2 = (j + 16 + dof + r > q) ? -__builtin_inff() : v2;
      }
      e[r] = v1; e[r + 8] = v2;
      rmax = fmaxf(rmax, fmaxf(v1, v2));
    }
    rmax = fmaxf(rmax, swap16(rmax));          // combine row halves (lane^16)
    float mnew  = fmaxf(m, rmax);
    float alpha = __builtin_amdgcn_exp2f(m - mnew);
    float lsum  = 0.f;
#pragma unroll
    for (int i = 0; i < 16; ++i) {
      e[i] = __builtin_amdgcn_exp2f(e[i] - mnew);
      lsum += e[i];
    }
    lsum += swap16(lsum);
    l = l * alpha + lsum;
    m = mnew;

    // P already sits in A-operand layout: just convert to bf16
    v16us pus;
#pragma unroll
    for (int i = 0; i < 16; ++i) pus[i] = f2bf(e[i]);
    v16bf pA = __builtin_bit_cast(v16bf, pus);

    // rescale O accumulators by alpha of each output row (r + 8*hi)
#pragma unroll
    for (int r = 0; r < 8; ++r) {
      float ar = bcast_lane(alpha, r + (hi ? 8 : 0));
      o0[r] *= ar; o1[r] *= ar; o2[r] *= ar; o3[r] *= ar;
    }

    // V tiles as B operands from transposed V: keys j+8h..+7 and j+16+8h..+7 per lane (table k order)
    const size_t kro = (size_t)(j + dof);
    v16bf b0 = __builtin_bit_cast(v16bf, __builtin_shufflevector(*(const v8us*)(vbase + (size_t)(col)      * TT + kro), *(const v8us*)(vbase + (size_t)(col)      * TT + kro + 16), 0, 1, 2, 3, 4, 5, 6, 7, 8, 9, 10, 11, 12, 13, 14, 15));
    v16bf b1 = __builtin_bit_cast(v16bf, __builtin_shufflevector(*(const v8us*)(vbase + (size_t)(col + 16) * TT + kro), *(const v8us*)(vbase + (size_t)(col + 16) * TT + kro + 16), 0, 1, 2, 3, 4, 5, 6, 7, 8, 9, 10, 11, 12, 13, 14, 15));
    v16bf b2 = __builtin_bit_cast(v16bf, __builtin_shufflevector(*(const v8us*)(vbase + (size_t)(col + 32) * TT + kro), *(const v8us*)(vbase + (size_t)(col + 32) * TT + kro + 16), 0, 1, 2, 3, 4, 5, 6, 7, 8, 9, 10, 11, 12, 13, 14, 15));
    v16bf b3 = __builtin_bit_cast(v16bf, __builtin_shufflevector(*(const v8us*)(vbase + (size_t)(col + 48) * TT + kro), *(const v8us*)(vbase + (size_t)(col + 48) * TT + kro + 16), 0, 1, 2, 3, 4, 5, 6, 7, 8, 9, 10, 11, 12, 13, 14, 15));
    o0 = __builtin_amdgcn_wmma_f32_16x16x32_bf16(false, pA, false, b0, (short)0, o0, false, false);
    o1 = __builtin_amdgcn_wmma_f32_16x16x32_bf16(false, pA, false, b1, (short)0, o1, false, false);
    o2 = __builtin_amdgcn_wmma_f32_16x16x32_bf16(false, pA, false, b2, (short)0, o2, false, false);
    o3 = __builtin_amdgcn_wmma_f32_16x16x32_bf16(false, pA, false, b3, (short)0, o3, false, false);
  };

  // blocks [0, jFull) need no mask (j+31 <= q0 <= all queries);
  // exactly one masked tail block at jFull covers the causal diagonal.
  const int jFull = (q0 >= 31) ? (((q0 - 31) >> 5) << 5) + 32 : 0;
  for (int j = 0; j < jFull; j += 32) step(j, false);
  step(jFull, true);

  // epilogue: divide by row sums, write head_out[b,h,t,64] f32
  float* hp = hout + (size_t)bh * TT * CC;
#pragma unroll
  for (int r = 0; r < 8; ++r) {
    float lr  = bcast_lane(l, r + (hi ? 8 : 0));
    float inv = __builtin_amdgcn_rcpf(lr);
    int row = q0 + r + (hi ? 8 : 0);
    float* o = hp + (size_t)row * CC + col;
    o[0]  = o0[r] * inv;
    o[16] = o1[r] * inv;
    o[32] = o2[r] * inv;
    o[48] = o3[r] * inv;
  }
}

// ---------------- Phase 3: sum heads, apply Wo^T + bo -----------------------
__global__ __launch_bounds__(256) void out_kernel(
    const float* __restrict__ hout, const float* __restrict__ Wo,
    const float* __restrict__ bo, float* __restrict__ out) {
  int z = blockIdx.x * 256 + threadIdx.x;    // [0, B*T)
  int t = z & (TT - 1);
  int b = z >> 12;
  float de[CC];
#pragma unroll
  for (int c = 0; c < CC; ++c) de[c] = 0.f;
  for (int h = 0; h < HH; ++h) {
    const float* hp = hout + ((size_t)(b * HH + h) * TT + t) * CC;
#pragma unroll
    for (int c = 0; c < CC; ++c) de[c] += hp[c];
  }
  float* op = out + (size_t)z * CC;
  for (int co = 0; co < CC; ++co) {
    const float* w = Wo + co * CC;             // uniform -> scalar loads
    float acc = bo[co];
#pragma unroll
    for (int c = 0; c < CC; ++c) acc += de[c] * w[c];
    op[co] = acc;
  }
}

extern "C" void kernel_launch(void* const* d_in, const int* in_sizes, int n_in,
                              void* d_out, int out_size, void* d_ws, size_t ws_size,
                              hipStream_t stream) {
  const float* x  = (const float*)d_in[0];
  const float* Wq = (const float*)d_in[1];
  const float* Wk = (const float*)d_in[2];
  const float* Wv = (const float*)d_in[3];
  const float* Wo = (const float*)d_in[4];
  const float* bo = (const float*)d_in[5];
  float* out = (float*)d_out;

  char* ws = (char*)d_ws;
  unsigned short* qb = (unsigned short*)(ws);                            // 2 MB
  unsigned short* kb = (unsigned short*)(ws + (size_t)2  * 1024 * 1024); // 2 MB
  unsigned short* vt = (unsigned short*)(ws + (size_t)4  * 1024 * 1024); // 8 MB
  float*          hh = (float*)        (ws + (size_t)12 * 1024 * 1024);  // 16 MB

  proj_kernel<<<dim3((BB * HH * TT) / 256), 256, 0, stream>>>(x, Wq, Wk, Wv, qb, kb, vt);
  fa_kernel<<<dim3(TT / 128, HH, BB), 256, 0, stream>>>(qb, kb, vt, hh);
  out_kernel<<<dim3((BB * TT) / 256), 256, 0, stream>>>(hh, Wo, bo, out);
}
